// MSVED_74818330296465
// MI455X (gfx1250) — compile-verified, matrix operand order checked by reading, not run on hardware
//
#include <hip/hip_runtime.h>
#include <hip/hip_bf16.h>

// Problem dims
#define S_   512
#define B_   64
#define V_   2000
#define VP_  2048            // V padded to multiple of 32 (zero-filled)
#define H_   256
#define G3_  768             // 3*H
#define Z_   64
#define M_   (S_ * B_)       // 32768 rows of the big input-projection GEMM

typedef __attribute__((ext_vector_type(16))) __bf16 v16bf;
typedef __attribute__((ext_vector_type(8)))  __bf16 v8bf;
typedef __attribute__((ext_vector_type(8)))  float  v8f;
typedef __attribute__((ext_vector_type(4)))  unsigned int v4u;
typedef __attribute__((ext_vector_type(8)))  int v8i;
typedef __attribute__((ext_vector_type(4)))  int v4i;

#if __has_builtin(__builtin_amdgcn_tensor_load_to_lds) && __has_builtin(__builtin_amdgcn_s_wait_tensorcnt)
#define HAS_TDM 1
#else
#define HAS_TDM 0
#endif

union V16U { v16bf v; v8bf h[2]; };

// A/B fragment loader per CDNA5 16-bit WMMA layout:
// lanes 0-15 hold K {base+0..7, base+16..23}; lanes 16-31 hold K {base+8..15, base+24..31}.
// Caller passes p already offset by (lane<16 ? 0 : 8).
static __device__ inline v16bf load_frag(const __bf16* p) {
    V16U u;
    u.h[0] = *reinterpret_cast<const v8bf*>(p);
    u.h[1] = *reinterpret_cast<const v8bf*>(p + 16);
    return u.v;
}

#if HAS_TDM
// Issue one TDM 2D tile load: tile = 64 (contiguous K, dim0) x 128 rows (dim1),
// bf16 elements, row stride 2048 elements. D# per CDNA5 ISA ch.8.
static __device__ inline void tdm_load_2d(unsigned int lds_off, const void* gptr,
                                          unsigned int tensor_rows) {
    const unsigned long long ga = (unsigned long long)gptr;
    v4u g0;
    g0[0] = 1u;                                            // count=1, user descriptor
    g0[1] = lds_off;                                       // lds_addr (bytes)
    g0[2] = (unsigned int)(ga & 0xFFFFFFFFu);              // global_addr[31:0]
    g0[3] = (unsigned int)((ga >> 32) & 0x1FFFFFFu)        // global_addr[56:32]
          | (2u << 30);                                    // type = 2 ("image")
    v8i g1;
    g1[0] = (int)(1u << 16);                               // data_size=1 -> 2 bytes
    g1[1] = (int)((unsigned)VP_ << 16);                    // tensor_dim0[15:0]=2048
    g1[2] = (int)((tensor_rows & 0xFFFFu) << 16);          // dim0 hi=0 | tensor_dim1 lo
    g1[3] = (int)((tensor_rows >> 16) & 0xFFFFu)           // tensor_dim1 hi
          | (int)(64u << 16);                              // tile_dim0 = 64
    g1[4] = (int)128;                                      // tile_dim1=128, tile_dim2=0
    g1[5] = (int)VP_;                                      // tensor_dim0_stride lo32 = 2048
    g1[6] = 0;                                             // stride hi | dim1_stride lo
    g1[7] = 0;
    const v4i z4 = {0, 0, 0, 0};
    const v8i z8 = {0, 0, 0, 0, 0, 0, 0, 0};
    __builtin_amdgcn_tensor_load_to_lds(g0, g1, z4, z4, z8, 0);
}
#endif

// ---------------------------------------------------------------------------
// fp32 -> bf16 convert with K-padding (zero fill for k >= sk)
// ---------------------------------------------------------------------------
__global__ void cvt_pad_bf16(const float* __restrict__ src, __bf16* __restrict__ dst,
                             int rows, int sk, int dk) {
    long long i = (long long)blockIdx.x * blockDim.x + threadIdx.x;
    long long n = (long long)rows * dk;
    if (i >= n) return;
    int r = (int)(i / dk);
    int k = (int)(i % dk);
    float v = (k < sk) ? src[(long long)r * sk + k] : 0.0f;
    dst[i] = (__bf16)v;
}

// ---------------------------------------------------------------------------
// Big input-projection GEMM: Gi[dir] = X[M,VP] * W[dir][G3,VP]^T + bih[dir]
// WG = 256 threads = 8 waves, tile 128(M) x 128(N), wave tile 32x64.
// K is consumed in 64-wide chunks staged into LDS by the Tensor Data Mover
// (double buffered, TENSORcnt-synced); waves read fragments via ds_load_b128
// and run v_wmma_f32_16x16x32_bf16.
// LDS: A 2x[128][64] bf16 (2x16KB) + B 2x[128][64] bf16 (2x16KB) = 64KB.
// ---------------------------------------------------------------------------
__global__ __launch_bounds__(256) void gemm_gi(
    const __bf16* __restrict__ X,
    const __bf16* __restrict__ Wf, const __bf16* __restrict__ Wb,
    const float* __restrict__ bihf, const float* __restrict__ bihb,
    float* __restrict__ Gif, float* __restrict__ Gib) {
    const int dir = blockIdx.z;
    const __bf16* W  = dir ? Wb   : Wf;
    const float* bih = dir ? bihb : bihf;
    float*       Gi  = dir ? Gib  : Gif;

    extern __shared__ char smem[];
    __bf16* As = (__bf16*)smem;                 // [2][128][64] at byte 0 / 16384
    __bf16* Bs = (__bf16*)(smem + 32768);       // [2][128][64] at byte 32768 / 49152

    const int lane  = threadIdx.x & 31;
    const int wave  = threadIdx.x >> 5;
    const int wm    = wave & 3;          // 4 wave-rows
    const int wn    = wave >> 2;         // 2 wave-cols
    const int mblk  = blockIdx.x * 128;
    const int nblk  = blockIdx.y * 128;
    const int lrow  = lane & 15;
    const int khalf = (lane < 16) ? 0 : 8;
    const int NCH   = VP_ / 64;          // 32 K-chunks

#if HAS_TDM
    if (wave == 0) {
        tdm_load_2d(0u,      X + (size_t)mblk * VP_, (unsigned)M_);
        tdm_load_2d(32768u,  W + (size_t)nblk * VP_, (unsigned)G3_);
    }
#endif

    v8f acc[2][4] = {};

    for (int c = 0; c < NCH; ++c) {
        const int buf = c & 1;
#if HAS_TDM
        if (wave == 0) __builtin_amdgcn_s_wait_tensorcnt((unsigned short)0);
        __syncthreads();
        if (c + 1 < NCH && wave == 0) {
            const unsigned int nb = (unsigned)((c + 1) & 1);
            tdm_load_2d(nb * 16384u,          X + (size_t)mblk * VP_ + (c + 1) * 64, (unsigned)M_);
            tdm_load_2d(32768u + nb * 16384u, W + (size_t)nblk * VP_ + (c + 1) * 64, (unsigned)G3_);
        }
#else
        // Fallback: cooperative global->LDS staging (single buffer semantics).
        __syncthreads();
        {
            // 128 rows x 64 bf16 = 1024 chunks of 16B per tile; 256 threads x 4.
            for (int q = threadIdx.x; q < 1024; q += 256) {
                const int row = q >> 3;
                const int co  = (q & 7) * 8;
                *(v8bf*)(As + buf * 8192 + row * 64 + co) =
                    *(const v8bf*)(X + (size_t)(mblk + row) * VP_ + c * 64 + co);
                *(v8bf*)(Bs + buf * 8192 + row * 64 + co) =
                    *(const v8bf*)(W + (size_t)(nblk + row) * VP_ + c * 64 + co);
            }
        }
        __syncthreads();
#endif
        const __bf16* Ab = As + buf * 8192;
        const __bf16* Bb = Bs + buf * 8192;
#pragma unroll
        for (int ks = 0; ks < 64; ks += 32) {
            v16bf a[2], b[4];
#pragma unroll
            for (int mi = 0; mi < 2; ++mi)
                a[mi] = load_frag(Ab + (wm * 32 + mi * 16 + lrow) * 64 + ks + khalf);
#pragma unroll
            for (int ni = 0; ni < 4; ++ni)
                b[ni] = load_frag(Bb + (wn * 64 + ni * 16 + lrow) * 64 + ks + khalf);
#pragma unroll
            for (int mi = 0; mi < 2; ++mi)
#pragma unroll
                for (int ni = 0; ni < 4; ++ni)
                    acc[mi][ni] = __builtin_amdgcn_wmma_f32_16x16x32_bf16(
                        false, a[mi], false, b[ni], (short)0, acc[mi][ni], false, false);
        }
        __syncthreads();   // all waves done reading buf before TDM may overwrite it
    }

    // D layout: VGPR r -> M = r (lanes 0-15) / r+8 (lanes 16-31); N = lane&15.
    const int m0 = mblk + wm * 32;
    const int n0 = nblk + wn * 64;
#pragma unroll
    for (int mi = 0; mi < 2; ++mi)
#pragma unroll
        for (int ni = 0; ni < 4; ++ni) {
            const int col = n0 + ni * 16 + lrow;
            const float bv = bih[col];
#pragma unroll
            for (int r = 0; r < 8; ++r) {
                const int row = m0 + mi * 16 + ((lane < 16) ? r : r + 8);
                Gi[(size_t)row * G3_ + col] = acc[mi][ni][r] + bv;
            }
        }
}

// ---------------------------------------------------------------------------
// Persistent GRU recurrence, one workgroup per direction.
// LDS: GH[64][768] f32 (192KB) + h[64][256] f32 (64KB) + h_bf[64][256] (32KB).
// 32 wave32 waves; wave owns 2(M)x3(N) 16x16 tiles of gh = h_bf @ Whh_bf^T.
// ---------------------------------------------------------------------------
__global__ __launch_bounds__(1024) void gru_recur(
    const float* __restrict__ Gif, const float* __restrict__ Gib,
    const __bf16* __restrict__ Whhf, const __bf16* __restrict__ Whhb,
    const float* __restrict__ bhhf, const float* __restrict__ bhhb,
    float* __restrict__ u) {
    const int dir = blockIdx.x;
    const float*  Gi  = dir ? Gib  : Gif;
    const __bf16* Whh = dir ? Whhb : Whhf;
    const float*  bhh = dir ? bhhb : bhhf;

    extern __shared__ char smem[];
    float*  GH = (float*)smem;                                              // [64][768]
    float*  hS = (float*)(smem + (size_t)B_ * G3_ * 4);                     // [64][256]
    __bf16* hB = (__bf16*)(smem + (size_t)B_ * G3_ * 4 + (size_t)B_ * H_ * 4); // [64][256]

    const int tid   = threadIdx.x;
    const int lane  = tid & 31;
    const int wave  = tid >> 5;
    const int wn    = wave & 15;         // 16 groups * 3 N-tiles = 48 N-tiles
    const int wm2   = wave >> 4;         // 2 groups  * 2 M-tiles = 4 M-tiles
    const int lrow  = lane & 15;
    const int khalf = (lane < 16) ? 0 : 8;

    for (int i = tid; i < B_ * H_; i += 1024) { hS[i] = 0.0f; hB[i] = (__bf16)0.0f; }
    __syncthreads();

    for (int t = 0; t < S_; ++t) {
        v8f acc[2][3] = {};
#pragma unroll
        for (int k0 = 0; k0 < H_; k0 += 32) {
            v16bf a[2], b[3];
#pragma unroll
            for (int mi = 0; mi < 2; ++mi) {
                const int row = (wm2 * 2 + mi) * 16 + lrow;
                a[mi] = load_frag(hB + row * H_ + k0 + khalf);
            }
#pragma unroll
            for (int ni = 0; ni < 3; ++ni) {
                const int col = (wn * 3 + ni) * 16 + lrow;
                b[ni] = load_frag(Whh + (size_t)col * H_ + k0 + khalf);
            }
#pragma unroll
            for (int mi = 0; mi < 2; ++mi)
#pragma unroll
                for (int ni = 0; ni < 3; ++ni)
                    acc[mi][ni] = __builtin_amdgcn_wmma_f32_16x16x32_bf16(
                        false, a[mi], false, b[ni], (short)0, acc[mi][ni], false, false);
        }
        // gh tiles -> LDS
#pragma unroll
        for (int mi = 0; mi < 2; ++mi)
#pragma unroll
            for (int ni = 0; ni < 3; ++ni) {
                const int n = (wn * 3 + ni) * 16 + lrow;
#pragma unroll
                for (int r = 0; r < 8; ++r) {
                    const int m = (wm2 * 2 + mi) * 16 + ((lane < 16) ? r : r + 8);
                    GH[m * G3_ + n] = acc[mi][ni][r];
                }
            }
        __syncthreads();

        // fused GRU gates (PyTorch GRUCell math); bih already folded into Gi
        const int trow = dir ? (S_ - 1 - t) : t;
        for (int i = tid; i < B_ * H_; i += 1024) {
            const int b = i >> 8;        // / 256
            const int j = i & 255;
            const float* gir = Gi + ((size_t)trow * B_ + b) * G3_;
            float r  = gir[j]           + GH[b * G3_ + j]           + bhh[j];
            float z  = gir[j + H_]      + GH[b * G3_ + j + H_]      + bhh[j + H_];
            float hn = GH[b * G3_ + j + 2 * H_] + bhh[j + 2 * H_];
            r = 1.0f / (1.0f + __expf(-r));
            z = 1.0f / (1.0f + __expf(-z));
            const float n = tanhf(gir[j + 2 * H_] + r * hn);
            const float hnew = (1.0f - z) * n + z * hS[i];
            hS[i] = hnew;
            hB[i] = (__bf16)hnew;
        }
        __syncthreads();
    }

    for (int i = tid; i < B_ * H_; i += 1024) {
        const int b = i >> 8, j = i & 255;
        u[(size_t)b * (2 * H_) + dir * H_ + j] = hS[i];
    }
}

// ---------------------------------------------------------------------------
// VAE head: mu=relu(u@Wmu^T+bmu), lv=relu(u@Wlv^T+blv), out=mu+eps*exp(0.5*lv)
// ---------------------------------------------------------------------------
__global__ void vae_head(const float* __restrict__ u,
                         const float* __restrict__ Wmu, const float* __restrict__ bmu,
                         const float* __restrict__ Wlv, const float* __restrict__ blv,
                         const float* __restrict__ eps, float* __restrict__ out) {
    const int i = blockIdx.x * blockDim.x + threadIdx.x;   // 0..4095
    if (i >= B_ * Z_) return;
    const int b = i >> 6, z = i & 63;
    const float* ur = u + (size_t)b * (2 * H_);
    float smu = bmu[z], slv = blv[z];
    const float* wm = Wmu + (size_t)z * (2 * H_);
    const float* wl = Wlv + (size_t)z * (2 * H_);
    for (int k = 0; k < 2 * H_; ++k) {
        const float uv = ur[k];
        smu = fmaf(uv, wm[k], smu);
        slv = fmaf(uv, wl[k], slv);
    }
    smu = fmaxf(smu, 0.0f);
    slv = fmaxf(slv, 0.0f);
    out[i] = smu + eps[i] * __expf(0.5f * slv);
}

// ---------------------------------------------------------------------------
extern "C" void kernel_launch(void* const* d_in, const int* in_sizes, int n_in,
                              void* d_out, int out_size, void* d_ws, size_t ws_size,
                              hipStream_t stream) {
    const float* x_s   = (const float*)d_in[0];
    const float* Wih_f = (const float*)d_in[1];
    const float* Whh_f = (const float*)d_in[2];
    const float* bih_f = (const float*)d_in[3];
    const float* bhh_f = (const float*)d_in[4];
    const float* Wih_b = (const float*)d_in[5];
    const float* Whh_b = (const float*)d_in[6];
    const float* bih_b = (const float*)d_in[7];
    const float* bhh_b = (const float*)d_in[8];
    const float* W_mu  = (const float*)d_in[9];
    const float* b_mu  = (const float*)d_in[10];
    const float* W_lv  = (const float*)d_in[11];
    const float* b_lv  = (const float*)d_in[12];
    const float* eps   = (const float*)d_in[13];

    char* ws = (char*)d_ws;
    size_t off = 0;
    __bf16* Xbf    = (__bf16*)(ws + off); off += (size_t)M_  * VP_ * 2;   // 134 MB
    __bf16* Wfbf   = (__bf16*)(ws + off); off += (size_t)G3_ * VP_ * 2;   // 3.1 MB
    __bf16* Wbbf   = (__bf16*)(ws + off); off += (size_t)G3_ * VP_ * 2;   // 3.1 MB
    __bf16* Whhfbf = (__bf16*)(ws + off); off += (size_t)G3_ * H_  * 2;   // 0.4 MB
    __bf16* Whhbbf = (__bf16*)(ws + off); off += (size_t)G3_ * H_  * 2;   // 0.4 MB
    float*  Gif    = (float*) (ws + off); off += (size_t)M_  * G3_ * 4;   // 100 MB
    float*  Gib    = (float*) (ws + off); off += (size_t)M_  * G3_ * 4;   // 100 MB
    float*  U      = (float*) (ws + off); off += (size_t)B_  * 2 * H_ * 4;

    // 1) precision/pad conversion (bf16, K padded to 2048 with zeros)
    {
        const long long nX = (long long)M_ * VP_;
        cvt_pad_bf16<<<(unsigned)((nX + 255) / 256), 256, 0, stream>>>(x_s, Xbf, M_, V_, VP_);
        const long long nW = (long long)G3_ * VP_;
        cvt_pad_bf16<<<(unsigned)((nW + 255) / 256), 256, 0, stream>>>(Wih_f, Wfbf, G3_, V_, VP_);
        cvt_pad_bf16<<<(unsigned)((nW + 255) / 256), 256, 0, stream>>>(Wih_b, Wbbf, G3_, V_, VP_);
        const long long nH = (long long)G3_ * H_;
        cvt_pad_bf16<<<(unsigned)((nH + 255) / 256), 256, 0, stream>>>(Whh_f, Whhfbf, G3_, H_, H_);
        cvt_pad_bf16<<<(unsigned)((nH + 255) / 256), 256, 0, stream>>>(Whh_b, Whhbbf, G3_, H_, H_);
    }

    // 2) hoisted input projections, both directions: TDM-staged WMMA GEMM, 64KB LDS
    gemm_gi<<<dim3(M_ / 128, G3_ / 128, 2), 256, 65536, stream>>>(
        Xbf, Wfbf, Wbbf, bih_f, bih_b, Gif, Gib);

    // 3) sequential recurrence: 1 persistent WG per direction, 288KB LDS
    {
        const size_t lds = (size_t)B_ * G3_ * 4 + (size_t)B_ * H_ * 4 + (size_t)B_ * H_ * 2;
        gru_recur<<<2, 1024, lds, stream>>>(Gif, Gib, Whhfbf, Whhbbf, bhh_f, bhh_b, U);
    }

    // 4) VAE head
    vae_head<<<(B_ * Z_ + 255) / 256, 256, 0, stream>>>(U, W_mu, b_mu, W_lv, b_lv, eps,
                                                        (float*)d_out);
}
